// SerialPatchEmbed_5385888989214
// MI455X (gfx1250) — matrix operand order repaired and checked by reading, not compiled, not run on hardware
//
#include <hip/hip_runtime.h>
#include <hip/hip_fp16.h>

// ---------------- problem constants (match reference setup_inputs) ---------
#define B_   16
#define N_   16384
#define GS_  16
#define G_   (N_ / GS_)      // 1024 groups per batch
#define HID_ 768
#define EMB_ 384

typedef __attribute__((ext_vector_type(16))) _Float16 v16h;
typedef __attribute__((ext_vector_type(8)))  _Float16 v8h;
typedef __attribute__((ext_vector_type(8)))  float    v8f;

__device__ __forceinline__ float gelu_exact(float x) {
    return 0.5f * x * (1.0f + erff(x * 0.70710678118654752f));
}

// -------------------------------------------------------------------------
// Kernel 1: per-batch bounding box (min/max over 16384 points)
// -------------------------------------------------------------------------
__global__ void bbox_kernel(const float* __restrict__ xyz,
                            float* __restrict__ bmin, float* __restrict__ bmax) {
    __shared__ float smin[3][256];
    __shared__ float smax[3][256];
    const int b = blockIdx.x, t = threadIdx.x;
    float mn[3] = { 3.4e38f, 3.4e38f, 3.4e38f };
    float mx[3] = { -3.4e38f, -3.4e38f, -3.4e38f };
    const float* base = xyz + (size_t)b * N_ * 3;
    for (int n = t; n < N_; n += 256) {
        #pragma unroll
        for (int c = 0; c < 3; ++c) {
            float v = base[(size_t)n * 3 + c];
            mn[c] = fminf(mn[c], v);
            mx[c] = fmaxf(mx[c], v);
        }
    }
    #pragma unroll
    for (int c = 0; c < 3; ++c) { smin[c][t] = mn[c]; smax[c][t] = mx[c]; }
    __syncthreads();
    for (int s = 128; s > 0; s >>= 1) {
        if (t < s) {
            #pragma unroll
            for (int c = 0; c < 3; ++c) {
                smin[c][t] = fminf(smin[c][t], smin[c][t + s]);
                smax[c][t] = fmaxf(smax[c][t], smax[c][t + s]);
            }
        }
        __syncthreads();
    }
    if (t == 0) {
        #pragma unroll
        for (int c = 0; c < 3; ++c) {
            bmin[b * 3 + c] = smin[c][0];
            bmax[b * 3 + c] = smax[c][0];
        }
    }
}

// -------------------------------------------------------------------------
// Kernel 2: Morton codes -> stable 64-bit sort keys (code<<14)|idx
// -------------------------------------------------------------------------
__device__ __forceinline__ unsigned part1by2_10(unsigned v) {
    v &= 1023u;
    v = (v | (v << 16)) & 0x030000FFu;
    v = (v | (v << 8))  & 0x0300F00Fu;
    v = (v | (v << 4))  & 0x030C30C3u;
    v = (v | (v << 2))  & 0x09249249u;
    return v;
}

__global__ void morton_kernel(const float* __restrict__ xyz,
                              const float* __restrict__ bmin,
                              const float* __restrict__ bmax,
                              unsigned long long* __restrict__ keys) {
    const int i = blockIdx.x * 256 + threadIdx.x;   // over B_*N_
    const int b = i / N_, n = i % N_;
    unsigned q[3];
    #pragma unroll
    for (int c = 0; c < 3; ++c) {
        float mn = bmin[b * 3 + c];
        float sc = fmaxf(bmax[b * 3 + c] - mn, 1e-6f);
        float u  = (xyz[(size_t)i * 3 + c] - mn) / sc;
        float r  = rintf(u * 1023.0f);              // round-half-even, matches jnp.round
        r = fminf(fmaxf(r, 0.0f), 1023.0f);
        q[c] = (unsigned)r;
    }
    unsigned code = part1by2_10(q[0]) | (part1by2_10(q[1]) << 1) | (part1by2_10(q[2]) << 2);
    keys[i] = ((unsigned long long)code << 14) | (unsigned)n;   // stable ascending
}

// -------------------------------------------------------------------------
// Kernel 3: full per-batch bitonic sort in LDS (16384 x u64 = 128 KB,
// fits CDNA5's 320 KB LDS), then gather xyz and emit group_idx.
// One 1024-thread block per batch.
// -------------------------------------------------------------------------
__global__ void sort_gather_kernel(const float* __restrict__ xyz,
                                   const unsigned long long* __restrict__ keys,
                                   float* __restrict__ xyz_s,
                                   int* __restrict__ out_gidx) {
    extern __shared__ char smem_raw[];
    unsigned long long* k = (unsigned long long*)smem_raw;
    const int b = blockIdx.x, t = threadIdx.x;
    const unsigned long long* kin = keys + (size_t)b * N_;
    for (int i = t; i < N_; i += 1024) k[i] = kin[i];
    __syncthreads();
    for (int kk = 2; kk <= N_; kk <<= 1) {
        for (int j = kk >> 1; j > 0; j >>= 1) {
            for (int i = t; i < N_; i += 1024) {
                int ixj = i ^ j;
                if (ixj > i) {
                    bool up = ((i & kk) == 0);
                    unsigned long long a = k[i], c = k[ixj];
                    if (up ? (a > c) : (a < c)) { k[i] = c; k[ixj] = a; }
                }
            }
            __syncthreads();
        }
    }
    for (int i = t; i < N_; i += 1024) {
        int idx = (int)(k[i] & 0x3FFFull);
        out_gidx[(size_t)b * N_ + i] = idx;
        size_t dst = ((size_t)b * N_ + i) * 3;
        size_t src = ((size_t)b * N_ + idx) * 3;
        xyz_s[dst + 0] = xyz[src + 0];
        xyz_s[dst + 1] = xyz[src + 1];
        xyz_s[dst + 2] = xyz[src + 2];
    }
}

// -------------------------------------------------------------------------
// Kernel 4: group centers (mean of 16 sorted points)
// -------------------------------------------------------------------------
__global__ void centers_kernel(const float* __restrict__ xyz_s,
                               float* __restrict__ cent_ws,
                               float* __restrict__ out_cent) {
    const int t = blockIdx.x * 256 + threadIdx.x;   // over B_*G_ == 16384
    float s0 = 0.f, s1 = 0.f, s2 = 0.f;
    const float* p = xyz_s + (size_t)t * GS_ * 3;
    #pragma unroll
    for (int i = 0; i < GS_; ++i) {
        s0 += p[i * 3 + 0]; s1 += p[i * 3 + 1]; s2 += p[i * 3 + 2];
    }
    const float inv = 1.0f / (float)GS_;
    s0 *= inv; s1 *= inv; s2 *= inv;
    cent_ws[t * 3 + 0] = s0; out_cent[t * 3 + 0] = s0;
    cent_ws[t * 3 + 1] = s1; out_cent[t * 3 + 1] = s1;
    cent_ws[t * 3 + 2] = s2; out_cent[t * 3 + 2] = s2;
}

// -------------------------------------------------------------------------
// Kernel 5: W2 (HID x EMB, f32 row-major) -> W2t (EMB x HID, f16) so each
// WMMA B-fragment is two runs of 8 contiguous halves per lane.
// -------------------------------------------------------------------------
__global__ void convert_w2_kernel(const float* __restrict__ W2,
                                  _Float16* __restrict__ W2t) {
    const int t = blockIdx.x * 256 + threadIdx.x;   // over HID_*EMB_
    const int kd = t % HID_, e = t / HID_;
    W2t[t] = (_Float16)W2[(size_t)kd * EMB_ + e];   // W2t[e][k]
}

// -------------------------------------------------------------------------
// Kernel 6: main WMMA kernel. 8 groups (128 points) per 256-thread block.
// Phase 1: h1 = GELU(rel @ W1 + b1) -> f16 in LDS (128 x 776 halves).
// Phase 2: each of 8 waves owns one group (one 16-row M-tile) and runs the
// 16x768 @ 768x384 GEMM with v_wmma_f32_16x16x32_f16, fusing +b2, GELU,
// group max-pool (row-max of each D tile) and the center projection.
// -------------------------------------------------------------------------
__global__ __launch_bounds__(256) void patch_gemm_kernel(
    const float* __restrict__ xyz_s, const float* __restrict__ cent,
    const float* __restrict__ W1, const float* __restrict__ b1,
    const _Float16* __restrict__ W2t, const float* __restrict__ b2,
    const float* __restrict__ Wc, const float* __restrict__ bc,
    float* __restrict__ out_tokens) {

    extern __shared__ char smem_raw[];
    _Float16* h1 = (_Float16*)smem_raw;             // [128][LDP]
    const int LDP = HID_ + 8;                       // pad to dodge bank conflicts

    const int tid = threadIdx.x;

    // ---- phase 1: first linear + GELU into LDS (f16) ----
    {
        const int p  = tid >> 1;                    // point 0..127 in block
        const int ks = (tid & 1) * (HID_ / 2);      // half of the 768 K values
        const size_t pg = (size_t)blockIdx.x * 128 + p;
        const int    cg = (int)(pg >> 4);           // group of this point
        const float rx = xyz_s[pg * 3 + 0] - cent[cg * 3 + 0];
        const float ry = xyz_s[pg * 3 + 1] - cent[cg * 3 + 1];
        const float rz = xyz_s[pg * 3 + 2] - cent[cg * 3 + 2];
        _Float16* dst = h1 + (size_t)p * LDP;
        for (int kv = ks; kv < ks + HID_ / 2; ++kv) {
            float h = fmaf(rx, W1[kv],
                      fmaf(ry, W1[HID_ + kv],
                      fmaf(rz, W1[2 * HID_ + kv], b1[kv])));
            dst[kv] = (_Float16)gelu_exact(h);
        }
    }
    __syncthreads();

    // ---- phase 2: WMMA GEMM + fused epilogue ----
    const int w    = tid >> 5;                      // wave id = group within block
    const int lane = tid & 31;
    const int lm   = lane & 15;
    const int khi  = lane >> 4;
    const int bg   = blockIdx.x * 8 + w;            // global group 0..16383
    const int row  = w * 16 + lm;                   // A-matrix row (M)
    const _Float16* Arow = h1 + (size_t)row * LDP;

    const float cx = cent[bg * 3 + 0];
    const float cy = cent[bg * 3 + 1];
    const float cz = cent[bg * 3 + 2];

    for (int nt = 0; nt < EMB_ / 16; ++nt) {
        const int col = nt * 16 + lm;               // output column (N)
        // B fragment base: lane holds column `col`, K rows khi*8+0..7 and 16+khi*8+0..7
        const _Float16* Bp = W2t + (size_t)col * HID_ + khi * 8;
        __builtin_prefetch(Bp, 0, 3);               // global_prefetch_b8

        v8f acc = {};
        #pragma unroll 4
        for (int kk = 0; kk < HID_ / 32; ++kk) {
            // A fragment (16-bit 16x32 layout): elems 0-7 -> K = khi*8+0..7,
            // elems 8-15 -> K = 16+khi*8+0..7 (within this 32-wide K slab)
            const _Float16* Ap = Arow + kk * 32 + khi * 8;
            v8h a0 = *(const v8h*)Ap;               // ds_load 16B
            v8h a1 = *(const v8h*)(Ap + 16);        // ds_load 16B
            v16h a = __builtin_shufflevector(a0, a1,
                        0, 1, 2, 3, 4, 5, 6, 7, 8, 9, 10, 11, 12, 13, 14, 15);

            const _Float16* Bk = Bp + kk * 32;
            v8h bl = *(const v8h*)Bk;               // global_load_b128
            v8h bh = *(const v8h*)(Bk + 16);
            v16h bb = __builtin_shufflevector(bl, bh,
                        0, 1, 2, 3, 4, 5, 6, 7, 8, 9, 10, 11, 12, 13, 14, 15);

            acc = __builtin_amdgcn_wmma_f32_16x16x32_f16(
                      false, a, false, bb, (short)0, acc, false, false);
        }

        // epilogue: +b2, exact GELU, max over the tile's 16 rows (= group max)
        const float bias = b2[col];
        float m = -3.4e38f;
        #pragma unroll
        for (int r = 0; r < 8; ++r) {               // rows M = r + 8*khi
            float v = gelu_exact(acc[r] + bias);
            m = fmaxf(m, v);
        }
        m = fmaxf(m, __shfl_xor(m, 16, 32));        // merge khi=0 / khi=1 halves
        if (lane < 16) {
            float cp = fmaf(cx, Wc[col],
                       fmaf(cy, Wc[EMB_ + col],
                       fmaf(cz, Wc[2 * EMB_ + col], bc[col])));
            out_tokens[(size_t)bg * EMB_ + col] = m + cp;
        }
    }
}

// -------------------------------------------------------------------------
// Host-side launcher
// -------------------------------------------------------------------------
extern "C" void kernel_launch(void* const* d_in, const int* in_sizes, int n_in,
                              void* d_out, int out_size, void* d_ws, size_t ws_size,
                              hipStream_t stream) {
    (void)in_sizes; (void)n_in; (void)out_size; (void)ws_size;

    const float* xyz = (const float*)d_in[0];
    const float* W1  = (const float*)d_in[1];
    const float* b1  = (const float*)d_in[2];
    const float* W2  = (const float*)d_in[3];
    const float* b2  = (const float*)d_in[4];
    const float* Wc  = (const float*)d_in[5];
    const float* bc  = (const float*)d_in[6];

    float* out        = (float*)d_out;
    float* out_tokens = out;                                        // B*G*EMB
    float* out_cent   = out + (size_t)B_ * G_ * EMB_;               // B*G*3
    int*   out_gidx   = (int*)(out + (size_t)B_ * G_ * EMB_ + (size_t)B_ * G_ * 3);

    // workspace layout (~6.03 MB total)
    char* ws = (char*)d_ws;
    float* bmin = (float*)(ws + 0);                                 // 16*3
    float* bmax = bmin + B_ * 3;                                    // 16*3
    unsigned long long* keys = (unsigned long long*)(ws + 512);     // 2 MB
    float* xyz_s  = (float*)(ws + 512 + 2097152);                   // 3 MB
    float* cent   = (float*)(ws + 512 + 2097152 + 3145728);         // 192 KB
    _Float16* W2t = (_Float16*)(ws + 512 + 2097152 + 3145728 + 196608); // 576 KB

    bbox_kernel<<<B_, 256, 0, stream>>>(xyz, bmin, bmax);
    morton_kernel<<<(B_ * N_) / 256, 256, 0, stream>>>(xyz, bmin, bmax, keys);
    sort_gather_kernel<<<B_, 1024, (size_t)N_ * 8, stream>>>(xyz, keys, xyz_s, out_gidx);
    centers_kernel<<<(B_ * G_) / 256, 256, 0, stream>>>(xyz_s, cent, out_cent);
    convert_w2_kernel<<<(HID_ * EMB_) / 256, 256, 0, stream>>>(W2, W2t);
    patch_gemm_kernel<<<(B_ * G_) / 8, 256, (size_t)128 * (HID_ + 8) * 2, stream>>>(
        xyz_s, cent, W1, b1, W2t, b2, Wc, bc, out_tokens);
}
